// RNN_82489141887677
// MI455X (gfx1250) — matrix operand order repaired and checked by reading, not compiled, not run on hardware
//
#include <hip/hip_runtime.h>
#include <hip/hip_bf16.h>
#include <stdint.h>

// Problem constants (reference: N=64, L=2048, V=32000, H=512)
#define NB 64
#define LL 2048
#define HH 512
#define NWG 16        // persistent workgroups in the scan phase

typedef __bf16 bf16;
typedef __attribute__((ext_vector_type(16))) bf16  v16bf;
typedef __attribute__((ext_vector_type(8)))  bf16  v8bf;
typedef __attribute__((ext_vector_type(8)))  float v8f;
typedef __attribute__((ext_vector_type(4)))  float v4f;
typedef __attribute__((ext_vector_type(4)))  unsigned v4u;
typedef __attribute__((ext_vector_type(8)))  int   v8i;
typedef __attribute__((ext_vector_type(4)))  int   v4i;

#if defined(__has_builtin)
#if __has_builtin(__builtin_amdgcn_tensor_load_to_lds) && __has_builtin(__builtin_amdgcn_s_wait_tensorcnt)
#define ATH_HAVE_TDM 1
#endif
#endif

// ---------------------------------------------------------------------------
// Workspace layout (all offsets 256B aligned):
//   [0,256)                     : grid-barrier counter
//   [256, +64KB)                : h ping  (64x512 bf16)
//   [.., +64KB)                 : h pong
//   [.., +512KB)                : W_xh bf16 (row-major [k_out][h])
//   [.., +512KB)                : W_hh bf16 (row-major [k_out][h])
// ---------------------------------------------------------------------------
#define WS_CNT 0
#define WS_H0  256
#define WS_H1  (WS_H0 + NB*HH*2)
#define WS_WXH (WS_H1 + NB*HH*2)
#define WS_WHH (WS_WXH + HH*HH*2)

// -------------------------- init: cvt weights, zero state -------------------
__global__ __launch_bounds__(256) void k_init(const float* __restrict__ Wxh,
                                              const float* __restrict__ Whh,
                                              bf16* wxh_bf, bf16* whh_bf,
                                              bf16* h0, bf16* h1, unsigned* counter) {
  int i = blockIdx.x * 256 + threadIdx.x;
  if (i < HH * HH) {
    wxh_bf[i] = (bf16)Wxh[i];
    whh_bf[i] = (bf16)Whh[i];
  }
  if (i < NB * HH) {
    h0[i] = (bf16)0.0f;
    h1[i] = (bf16)0.0f;
  }
  if (i == 0) *counter = 0u;
}

// -------------------------- phase 1: Xh = emb[X] @ Wxh^T + bxh --------------
// One wave computes one 16x16 output tile with K=512 (16 x wmma K=32).
// A (gathered embedding rows, f32) is converted to bf16 on the fly in the
// contiguous per-lane K layout (half h holds K 16h..16h+15) that B also uses.
__global__ __launch_bounds__(128) void k_xproj(const int* __restrict__ X,
                                               const float* __restrict__ emb,
                                               const bf16* __restrict__ wxh_bf,
                                               const float* __restrict__ bxh,
                                               float* __restrict__ out) {
  const int lane  = threadIdx.x & 31;
  const int wid   = blockIdx.x * 4 + (threadIdx.x >> 5);
  const int ntile = wid & 31;    // 32 col tiles
  const int mtile = wid >> 5;    // 8192 row tiles
  const int half  = lane >> 4;
  const int l15   = lane & 15;

  const int r = mtile * 16 + l15;                 // flat (n,l) row for this lane
  const int xrow = X[r];
  const float* arow = emb + (size_t)xrow * HH;    // gathered embedding row
  const int col = ntile * 16 + l15;
  const bf16* brow = wxh_bf + (size_t)col * HH;   // W_xh row = output column

  v8f acc = {};
#pragma unroll
  for (int c = 0; c < 16; ++c) {
    // A lane layout (16-bit 16x32): lanes 0-15 hold K 0..15, 16-31 hold K 16..31 (matches B)
    const int ka = c * 32 + half * 16;
    v4f f0 = *(const v4f*)(arow + ka);
    v4f f1 = *(const v4f*)(arow + ka + 4);
    v4f f2 = *(const v4f*)(arow + ka + 8);
    v4f f3 = *(const v4f*)(arow + ka + 12);
    v16bf a;
#pragma unroll
    for (int i = 0; i < 4; ++i) {
      a[i]      = (bf16)f0[i];
      a[i + 4]  = (bf16)f1[i];
      a[i + 8]  = (bf16)f2[i];
      a[i + 12] = (bf16)f3[i];
    }
    // B lane layout (32x16): lanes 0-15 hold K 0..15 of column `lane`, 16-31 hold K 16..31
    const int kb = c * 32 + half * 16;
    v16bf b = *(const v16bf*)(brow + kb);
    acc = __builtin_amdgcn_wmma_f32_16x16x32_bf16(false, a, false, b, (short)0, acc,
                                                  false, false);
  }
  const float bias = bxh[col];
  // C/D layout: VGPR v -> M = v + 8*half ; N = lane&15
#pragma unroll
  for (int v = 0; v < 8; ++v) {
    const int m = mtile * 16 + v + half * 8;
    out[(size_t)m * HH + col] = acc[v] + bias;
  }
}

// -------------------------- phase 2: recurrent scan -------------------------
// 16 persistent WGs x 4 waves. WG w owns output cols [32w, 32w+32).
// Wave v: N-tile = (v>>1), M rows [32*(v&1), +32) -> two wmma accumulate chains.
// W_hh slice staged into LDS with the Tensor Data Mover, then held in VGPRs.
__global__ __launch_bounds__(128, 1) void k_scan(const float* __restrict__ bhh,
                                                 const bf16* __restrict__ whh_bf,
                                                 bf16* h0, bf16* h1,
                                                 unsigned* counter,
                                                 float* __restrict__ out) {
  __shared__ bf16 lds_whh[32 * HH];  // 32KB: this WG's 32 W_hh rows
  const int w    = blockIdx.x;       // 0..15
  const int tid  = threadIdx.x;
  const int lane = tid & 31;
  const int wv   = tid >> 5;
  const int half = lane >> 4;
  const int l15  = lane & 15;

  const bf16* wsrc = whh_bf + (size_t)32 * HH * w;

#if defined(ATH_HAVE_TDM)
  if (wv == 0) {
    // D# group0: count=1, lds_addr, 57b global_addr, type=2
    const unsigned long long ga = (unsigned long long)(uintptr_t)wsrc;
    const unsigned lds_off = (unsigned)(uintptr_t)(&lds_whh[0]);  // low 32b of generic = LDS offset
    v4u g0;
    g0[0] = 1u;
    g0[1] = lds_off;
    g0[2] = (unsigned)(ga & 0xffffffffu);
    g0[3] = (unsigned)((ga >> 32) & 0x1ffffffu) | 0x80000000u;  // type=2 @ bits[127:126]
    // D# group1: data_size=2B; tensor 512x512; tile 512x32; dim0 stride 512
    v8i g1;
    g1[0] = 0x00010000;          // data_size=1 (2 bytes), wg_mask=0 (not in cluster)
    g1[1] = (int)(512u << 16);   // tensor_dim0[15:0] @ bits[63:48]
    g1[2] = (int)(512u << 16);   // tensor_dim1[15:0] @ bits[95:80]
    g1[3] = (int)(512u << 16);   // tile_dim0=512    @ bits[127:112]
    g1[4] = 32;                  // tile_dim1=32
    g1[5] = 512;                 // tensor_dim0_stride low32
    g1[6] = (int)(512u << 16);   // tensor_dim1_stride low16 (unused for 2D tile)
    g1[7] = 0;
    v4i gz = {0, 0, 0, 0};
#if __clang_major__ >= 23
    v8i gz8 = {0, 0, 0, 0, 0, 0, 0, 0};
    __builtin_amdgcn_tensor_load_to_lds(g0, g1, gz, gz, gz8, 0);
#else
    __builtin_amdgcn_tensor_load_to_lds(g0, g1, gz, gz, 0);
#endif
    __builtin_amdgcn_s_wait_tensorcnt(0);
  }
#else
  // Fallback: cooperative global->LDS copy (16B per thread-iteration)
  for (int i = tid; i < (32 * HH * 2) / 16; i += 128)
    ((uint4*)lds_whh)[i] = ((const uint4*)wsrc)[i];
#endif
  __syncthreads();

  const int nt = wv >> 1;                 // N-tile within WG (0/1)
  const int mp = wv & 1;                  // M-pair (rows 32*mp .. +32)
  const int colLocal = nt * 16 + l15;     // row within LDS slice
  const int col = 32 * w + colLocal;      // global output column

  // B operand (W_hh) resident in 128 VGPRs for the whole scan
  v16bf B[16];
#pragma unroll
  for (int c = 0; c < 16; ++c)
    B[c] = *(const v16bf*)(&lds_whh[colLocal * HH + c * 32 + half * 16]);

  const float bias = bhh[col];
  const int m0 = mp * 32 + l15;   // A-row (batch index), sub-tile 0
  const int m1 = m0 + 16;         // sub-tile 1

  bf16* hb0 = h0;
  bf16* hb1 = h1;

  for (int t = 0; t < LL; ++t) {
    const bf16* cur = (t & 1) ? hb1 : hb0;
    bf16* nxt       = (t & 1) ? hb0 : hb1;
    const bf16* arow0 = cur + (size_t)m0 * HH;
    const bf16* arow1 = cur + (size_t)m1 * HH;

    v8f acc0 = {}, acc1 = {};
#pragma unroll
    for (int c = 0; c < 16; ++c) {
      const int ka = c * 32 + half * 16;
      v8bf a0l = *(const v8bf*)(arow0 + ka);
      v8bf a0h = *(const v8bf*)(arow0 + ka + 8);
      v8bf a1l = *(const v8bf*)(arow1 + ka);
      v8bf a1h = *(const v8bf*)(arow1 + ka + 8);
      v16bf a0 = __builtin_shufflevector(a0l, a0h, 0, 1, 2, 3, 4, 5, 6, 7,
                                         8, 9, 10, 11, 12, 13, 14, 15);
      v16bf a1 = __builtin_shufflevector(a1l, a1h, 0, 1, 2, 3, 4, 5, 6, 7,
                                         8, 9, 10, 11, 12, 13, 14, 15);
      acc0 = __builtin_amdgcn_wmma_f32_16x16x32_bf16(false, a0, false, B[c], (short)0,
                                                     acc0, false, false);
      acc1 = __builtin_amdgcn_wmma_f32_16x16x32_bf16(false, a1, false, B[c], (short)0,
                                                     acc1, false, false);
    }

    // epilogue: h = tanh(acc + b_hh + Xh);  Xh lives in d_out[:,t,:], overwritten in place
#pragma unroll
    for (int v = 0; v < 8; ++v) {
      const int mm0 = mp * 32 + v + half * 8;
      const int mm1 = mm0 + 16;
      const size_t o0 = ((size_t)mm0 * LL + t) * HH + col;
      const size_t o1 = ((size_t)mm1 * LL + t) * HH + col;
      const float hv0 = tanhf(acc0[v] + bias + out[o0]);
      const float hv1 = tanhf(acc1[v] + bias + out[o1]);
      out[o0] = hv0;
      out[o1] = hv1;
      nxt[(size_t)mm0 * HH + col] = (bf16)hv0;
      nxt[(size_t)mm1 * HH + col] = (bf16)hv1;
      if (t + 1 < LL) {   // prefetch next step's Xh (global_prefetch_b8)
        __builtin_prefetch(&out[o0 + HH], 0, 1);
        __builtin_prefetch(&out[o1 + HH], 0, 1);
      }
    }

    // device-wide step barrier (monotone counter; reset each launch by k_init)
    __threadfence();
    __syncthreads();
    if (tid == 0) {
      __hip_atomic_fetch_add(counter, 1u, __ATOMIC_RELEASE, __HIP_MEMORY_SCOPE_AGENT);
      const unsigned target = (unsigned)(NWG * (t + 1));
      while (__hip_atomic_load(counter, __ATOMIC_ACQUIRE, __HIP_MEMORY_SCOPE_AGENT) < target)
        __builtin_amdgcn_s_sleep(1);
    }
    __syncthreads();
    __threadfence();
  }
}

// ---------------------------------------------------------------------------
extern "C" void kernel_launch(void* const* d_in, const int* in_sizes, int n_in,
                              void* d_out, int out_size, void* d_ws, size_t ws_size,
                              hipStream_t stream) {
  const int*   X   = (const int*)d_in[0];
  const float* emb = (const float*)d_in[1];
  const float* Whh = (const float*)d_in[2];
  const float* bhh = (const float*)d_in[3];
  const float* Wxh = (const float*)d_in[4];
  const float* bxh = (const float*)d_in[5];
  float* out = (float*)d_out;

  char* ws = (char*)d_ws;
  unsigned* counter = (unsigned*)(ws + WS_CNT);
  bf16* h0     = (bf16*)(ws + WS_H0);
  bf16* h1     = (bf16*)(ws + WS_H1);
  bf16* wxh_bf = (bf16*)(ws + WS_WXH);
  bf16* whh_bf = (bf16*)(ws + WS_WHH);

  // 1) convert weights to bf16, zero h state + barrier counter
  k_init<<<(HH * HH + 255) / 256, 256, 0, stream>>>(Wxh, Whh, wxh_bf, whh_bf, h0, h1,
                                                    counter);
  // 2) batched input projection into d_out (Xh)
  k_xproj<<<(NB * LL / 16) * (HH / 16) / 4, 128, 0, stream>>>(X, emb, wxh_bf, bxh, out);
  // 3) persistent recurrent scan, in-place over d_out
  k_scan<<<NWG, 128, 0, stream>>>(bhh, whh_bf, h0, h1, counter, out);
}
